// Experts_41429254537622
// MI455X (gfx1250) — matrix operand order repaired and checked by reading, not compiled, not run on hardware
//
#include <hip/hip_runtime.h>
#include <hip/hip_bf16.h>

// Problem constants (B=1, S=512, H=1024, I=2048, E=8, TOPK=2)
#define Hdim   1024
#define Idim   2048
#define TWO_I  4096
#define NTOK   1024          // B*S*TOPK routed rows
#define TOPK_  2
#define NEXP   8
#define MTILE  16
#define MAXT   (NTOK / MTILE + NEXP)   // 72 single-expert row tiles max

#define APAD     8
#define ASTRIDE  (Hdim + APAD)         // 1032 halfs -> 2064B rows, 16B aligned, bank-friendly
#define CHUNK    128
#define CPAD     8
#define CSTRIDE  (CHUNK + CPAD)        // 136 halfs -> 272B rows

typedef __attribute__((ext_vector_type(16))) __bf16 v16bf;
typedef __attribute__((ext_vector_type(8)))  __bf16 v8bf;
typedef __attribute__((ext_vector_type(4)))  __bf16 v4bf;
typedef __attribute__((ext_vector_type(8)))  float  v8f;
typedef __attribute__((ext_vector_type(4)))  float  v4f;

union ABu { v16bf v; v8bf h[2]; };

// ---------------------------------------------------------------------------
// Routing: stable counting sort by expert id + tile table (single block).
// ---------------------------------------------------------------------------
__global__ __launch_bounds__(256) void moe_route(const int* __restrict__ tpe,
                                                 int* __restrict__ perm_src,
                                                 int* __restrict__ tiles,
                                                 int* __restrict__ ntiles) {
  __shared__ int sflat[NTOK];
  __shared__ int sbase[NEXP];
  __shared__ int scnt[NEXP];
  const int tid = threadIdx.x;
  for (int i = tid; i < NTOK; i += 256) sflat[i] = tpe[i];
  if (tid < NEXP) scnt[tid] = 0;
  __syncthreads();
  for (int i = tid; i < NTOK; i += 256) atomicAdd(&scnt[sflat[i]], 1);
  __syncthreads();
  if (tid == 0) {
    int run = 0, nt = 0;
    for (int e = 0; e < NEXP; ++e) {
      sbase[e] = run;
      const int c = scnt[e];
      for (int off = 0; off < c; off += MTILE) {
        tiles[nt * 4 + 0] = e;
        tiles[nt * 4 + 1] = run + off;
        tiles[nt * 4 + 2] = (c - off < MTILE) ? (c - off) : MTILE;
        ++nt;
      }
      run += c;
    }
    *ntiles = nt;
  }
  __syncthreads();
  // Stable rank within expert == stable argsort semantics.
  for (int i = tid; i < NTOK; i += 256) {
    const int e = sflat[i];
    int rank = 0;
    for (int j = 0; j < i; ++j) rank += (sflat[j] == e) ? 1 : 0;
    perm_src[sbase[e] + rank] = i;
  }
}

// ---------------------------------------------------------------------------
// Fused grouped GEMM1 + SwiGLU + GEMM2 per 16-row single-expert tile.
// 8 waves/block; each wave owns a 16-col slice per chunk (GEMM1) and a
// persistent 16x128 output slice (GEMM2 accumulators in VGPRs).
// ---------------------------------------------------------------------------
__global__ __launch_bounds__(256) void moe_expert_kernel(
    const float*  __restrict__ x,
    const __bf16* __restrict__ w1,
    const __bf16* __restrict__ w2,
    const int*    __restrict__ perm_src,
    const int*    __restrict__ tiles,
    const int*    __restrict__ ntiles,
    __bf16*       __restrict__ out) {
  __shared__ __bf16 a_lds[MTILE * ASTRIDE];   // 33,024 B : bf16 activations, full K
  __shared__ __bf16 c_lds[MTILE * CSTRIDE];   //  4,352 B : current SwiGLU chunk
  __shared__ int    stok[MTILE];

  const int bid = blockIdx.x;
  if (bid >= *ntiles) return;
  const int e    = tiles[bid * 4 + 0];
  const int row0 = tiles[bid * 4 + 1];
  const int cnt  = tiles[bid * 4 + 2];

  const int tid  = threadIdx.x;
  const int lane = tid & 31;
  const int wave = tid >> 5;
  const int m    = lane & 15;   // output row within tile (A-frag row)
  const int kh   = lane >> 4;   // K-half selector per WMMA A layout

  if (tid < MTILE) {
    const int r = (tid < cnt) ? (row0 + tid) : row0;  // clamp: masked at store
    stok[tid] = perm_src[r] / TOPK_;
  }
  __syncthreads();

  // Phase A: gather 16 activation rows, fp32 -> bf16, into LDS (coalesced b128).
  for (int it = 0; it < (MTILE * Hdim) / (256 * 4); ++it) {   // 16 iters
    const int lin = (it * 256 + tid) * 4;
    const int r = lin >> 10;           // / Hdim
    const int c = lin & (Hdim - 1);
    const v4f f = *reinterpret_cast<const v4f*>(x + (size_t)stok[r] * Hdim + c);
    v4bf b4;
    b4[0] = (__bf16)f[0]; b4[1] = (__bf16)f[1];
    b4[2] = (__bf16)f[2]; b4[3] = (__bf16)f[3];
    *reinterpret_cast<v4bf*>(&a_lds[r * ASTRIDE + c]) = b4;
  }
  __syncthreads();

  const __bf16* w1e = w1 + (size_t)e * Hdim * TWO_I;
  const __bf16* w2e = w2 + (size_t)e * Idim * Hdim;

  const v8f vzero = {};
  v8f acc2[8];                        // 16 x 128 f32 output slice per wave
  #pragma unroll
  for (int t = 0; t < 8; ++t) acc2[t] = vzero;

  for (int chunk = 0; chunk < Idim / CHUNK; ++chunk) {   // 16 chunks of I
    // ---- GEMM1: inter cols [n0, n0+16) over full K = H, gate + up halves
    const int n0 = chunk * CHUNK + wave * 16;
    v8f accA = vzero, accB = vzero;
    #pragma unroll 2
    for (int k0 = 0; k0 < Hdim; k0 += 32) {
      ABu a;
      const __bf16* arow = &a_lds[m * ASTRIDE + k0 + kh * 8];
      a.h[0] = *reinterpret_cast<const v8bf*>(arow);
      a.h[1] = *reinterpret_cast<const v8bf*>(arow + 16);
      ABu ba, bb;                       // B frag: lane = N col, K packed in VGPRs
      const __bf16* brow = w1e + (size_t)(k0 + kh * 8) * TWO_I + n0 + m;
      #pragma unroll
      for (int s = 0; s < 16; ++s) {
        const int kk = (s < 8) ? s : (s + 8);
        ba.v[s] = brow[(size_t)kk * TWO_I];
        bb.v[s] = brow[(size_t)kk * TWO_I + Idim];
      }
      accA = __builtin_amdgcn_wmma_f32_16x16x32_bf16(false, a.v, false, ba.v,
                                                     (short)0, accA, false, false);
      accB = __builtin_amdgcn_wmma_f32_16x16x32_bf16(false, a.v, false, bb.v,
                                                     (short)0, accB, false, false);
    }

    __syncthreads();   // previous chunk's c_lds fully consumed
    {
      // C layout: VGPR r, lanes 0-15 -> M=r, lanes 16-31 -> M=r+8, N=lane&15
      const int ccol = wave * 16 + (lane & 15);
      #pragma unroll
      for (int r = 0; r < 8; ++r) {
        const int mm = r + kh * 8;
        const float av = accA[r];
        const float bv = accB[r];
        // silu(a)*b with fast v_rcp_f32 (matches jax.nn.silu accuracy at bf16)
        const float s = av * __builtin_amdgcn_rcpf(1.0f + __expf(-av));
        c_lds[mm * CSTRIDE + ccol] = (__bf16)(s * bv);
      }
    }
    __syncthreads();   // chunk of SwiGLU output visible to all waves

    // ---- GEMM2 partial: acc2 += inter_chunk[16x128] @ w2[k_chunk, wave cols]
    #pragma unroll
    for (int ks = 0; ks < CHUNK; ks += 32) {
      ABu a;
      const __bf16* arow = &c_lds[m * CSTRIDE + ks + kh * 8];
      a.h[0] = *reinterpret_cast<const v8bf*>(arow);
      a.h[1] = *reinterpret_cast<const v8bf*>(arow + 16);
      const int kglob = chunk * CHUNK + ks + kh * 8;
      const __bf16* b2row = w2e + (size_t)kglob * Hdim;
      #pragma unroll
      for (int t = 0; t < 8; ++t) {
        const int n2 = wave * 128 + t * 16;
        ABu b2;
        #pragma unroll
        for (int s = 0; s < 16; ++s) {
          const int kk = (s < 8) ? s : (s + 8);
          b2.v[s] = b2row[(size_t)kk * Hdim + n2 + m];
        }
        acc2[t] = __builtin_amdgcn_wmma_f32_16x16x32_bf16(false, a.v, false, b2.v,
                                                          (short)0, acc2[t], false, false);
      }
    }
  }

  // Store: bf16 output rows [row0, row0+cnt), wave's 128-col slice of H.
  #pragma unroll
  for (int t = 0; t < 8; ++t) {
    const int n2 = wave * 128 + t * 16 + (lane & 15);
    #pragma unroll
    for (int r = 0; r < 8; ++r) {
      const int mm = r + kh * 8;
      if (mm < cnt) {
        out[(size_t)(row0 + mm) * Hdim + n2] = (__bf16)acc2[t][r];
      }
    }
  }
}

// ---------------------------------------------------------------------------
extern "C" void kernel_launch(void* const* d_in, const int* in_sizes, int n_in,
                              void* d_out, int out_size, void* d_ws, size_t ws_size,
                              hipStream_t stream) {
  (void)in_sizes; (void)n_in; (void)out_size; (void)ws_size;
  const float*  x   = (const float*)d_in[0];
  const int*    tpe = (const int*)d_in[1];
  const __bf16* w1  = (const __bf16*)d_in[2];
  const __bf16* w2  = (const __bf16*)d_in[3];
  __bf16*       out = (__bf16*)d_out;

  int* perm_src = (int*)d_ws;           // [NTOK]
  int* tiles    = perm_src + NTOK;      // [MAXT*4]
  int* ntiles   = tiles + MAXT * 4;     // [1]

  moe_route<<<1, 256, 0, stream>>>(tpe, perm_src, tiles, ntiles);
  moe_expert_kernel<<<MAXT, 256, 0, stream>>>(x, w1, w2, perm_src, tiles, ntiles, out);
}
